// GeometricCliffordAttention_34892314313420
// MI455X (gfx1250) — matrix operand order repaired and checked by reading, not compiled, not run on hardware
//
#include <hip/hip_runtime.h>

#define B_ 4
#define S_ 2048
#define H_ 8
#define M_ 16

typedef _Float16 half_t;
typedef __attribute__((ext_vector_type(16))) _Float16 v16h;
typedef __attribute__((ext_vector_type(8)))  _Float16 v8h;
typedef __attribute__((ext_vector_type(8)))  float    v8f;

static __device__ __forceinline__ unsigned int pack_h2(float a, float b) {
    union { _Float16 h[2]; unsigned int u; } t;
    t.h[0] = (_Float16)a; t.h[1] = (_Float16)b;
    return t.u;
}

// ---------------------------------------------------------------------------
// Kernel 1: per-head Q/K/V projections (block-diagonal 16x16 linears).
// Writes Q (scaled by 1/4), K as f16 [b][h][s][d]; V as f16 TRANSPOSED [b][h][d][s]
// so the attention kernel's V^T A-operand loads are contiguous.
// ---------------------------------------------------------------------------
__global__ void proj_qkv(const float* __restrict__ x,
                         const float* __restrict__ wq, const float* __restrict__ bq,
                         const float* __restrict__ wk, const float* __restrict__ bk,
                         const float* __restrict__ wv, const float* __restrict__ bv,
                         half_t* __restrict__ Qh, half_t* __restrict__ Kh,
                         half_t* __restrict__ Vt)
{
    int t = blockIdx.x * blockDim.x + threadIdx.x;
    if (t >= H_ * B_ * S_) return;
    int h  = t / (B_ * S_);            // same h across a wave -> scalar weight loads
    int bs = t % (B_ * S_);
    int b  = bs / S_, s = bs % S_;

    float xv[16];
    const float* xp = x + (long)(b * S_ + s) * M_;
#pragma unroll
    for (int j = 0; j < 16; ++j) xv[j] = xp[j];

    float qa[16], ka[16], va[16];
#pragma unroll
    for (int i = 0; i < 16; ++i) {
        qa[i] = bq[h * M_ + i];
        ka[i] = bk[h * M_ + i];
        va[i] = bv[h * M_ + i];
    }
#pragma unroll
    for (int j = 0; j < 16; ++j) {
        const float* wqr = wq + (long)(j * H_ + h) * M_;
        const float* wkr = wk + (long)(j * H_ + h) * M_;
        const float* wvr = wv + (long)(j * H_ + h) * M_;
        float xj = xv[j];
#pragma unroll
        for (int i = 0; i < 16; ++i) {
            qa[i] = fmaf(xj, wqr[i], qa[i]);
            ka[i] = fmaf(xj, wkr[i], ka[i]);
            va[i] = fmaf(xj, wvr[i], va[i]);
        }
    }
    long base = ((long)(b * H_ + h) * S_ + s) * M_;
#pragma unroll
    for (int i = 0; i < 16; ++i) {
        Qh[base + i] = (half_t)(qa[i] * 0.25f);   // fold SCALE = 1/sqrt(16) into Q
        Kh[base + i] = (half_t)ka[i];
    }
    long vbase = (long)(b * H_ + h) * M_ * S_ + s;
#pragma unroll
    for (int i = 0; i < 16; ++i) Vt[vbase + (long)i * S_] = (half_t)va[i];
}

// ---------------------------------------------------------------------------
// Kernel 2: flash attention. One wave = one (b,h, 16-query) tile.
// Transposed formulation: S^T = K x Q^T, O^T = V^T x P^T, so softmax rows
// (queries) live one-per-lane and the P relayout is 8 xor-16 shuffles.
// ---------------------------------------------------------------------------
__global__ void __launch_bounds__(256)
flash_attn(const half_t* __restrict__ Qh, const half_t* __restrict__ Kh,
           const half_t* __restrict__ Vt, float* __restrict__ head)
{
    const int lane = threadIdx.x & 31;
    const int wave = threadIdx.x >> 5;
    const int gw   = blockIdx.x * (blockDim.x >> 5) + wave;   // 0 .. B*H*S/16-1
    const int qtile = gw & (S_ / 16 - 1);
    const int bh    = gw / (S_ / 16);
    const int q0    = qtile * 16;
    const int n     = lane & 15;
    const bool hi   = lane >= 16;

    const half_t* Qb = Qh + (long)bh * S_ * M_;
    const half_t* Kb = Kh + (long)bh * S_ * M_;
    const half_t* Vb = Vt + (long)bh * M_ * S_;

    // B operand for K x Q^T: Q^T (dim x query). Column/query n = lane&15;
    // lane<16: dims 0..7, lane>=16: dims 8..15 in elements 0..7; elements 8..15 = zero pad.
    v16h qf;
#pragma unroll
    for (int i = 0; i < 16; ++i) qf[i] = (_Float16)0.f;
    { v8h qh = *(const v8h*)(Qb + (long)(q0 + n) * M_ + (hi ? 8 : 0));
#pragma unroll
      for (int i = 0; i < 8; ++i) qf[i] = qh[i]; }

    v8f O;
#pragma unroll
    for (int r = 0; r < 8; ++r) O[r] = 0.f;
    float m = -1e30f, l = 0.f;

    const half_t* vrow = Vb + (long)n * S_ + (hi ? 16 : 0);

    for (int k0 = 0; k0 < S_; k0 += 32) {
        // ---- A operands for scores: K rows (M=key, K=dim padded to 32) ----
        // lane<16: row key=k0+n, d=0..7 ; lane>=16: row key=k0+n, d=8..15 ; halves 8..15 = 0
        v8h ka1 = *(const v8h*)(Kb + (long)(k0 + n) * M_ + (hi ? 8 : 0));
        v8h ka2 = *(const v8h*)(Kb + (long)(k0 + 16 + n) * M_ + (hi ? 8 : 0));
        v16h a1, a2;
#pragma unroll
        for (int i = 0; i < 8; ++i) {
            a1[i] = ka1[i]; a1[i + 8] = (_Float16)0.f;
            a2[i] = ka2[i]; a2[i + 8] = (_Float16)0.f;
        }
        v8f zc;
#pragma unroll
        for (int r = 0; r < 8; ++r) zc[r] = 0.f;

        // S^T tiles: D layout -> lane holds 8 keys x (query = lane&15)
        v8f s1 = __builtin_amdgcn_wmma_f32_16x16x32_f16(false, a1, false, qf,
                                                        (short)0, zc, false, false);
        v8f s2 = __builtin_amdgcn_wmma_f32_16x16x32_f16(false, a2, false, qf,
                                                        (short)0, zc, false, false);

        // ---- online softmax (per-query stats live per lane; xor-16 pairs agree) ----
        float tm = s1[0];
#pragma unroll
        for (int r = 1; r < 8; ++r) tm = fmaxf(tm, s1[r]);
#pragma unroll
        for (int r = 0; r < 8; ++r) tm = fmaxf(tm, s2[r]);
        tm = fmaxf(tm, __shfl_xor(tm, 16, 32));

        float mn    = fmaxf(m, tm);
        float alpha = __expf(m - mn);
        float p1[8], p2[8], rs = 0.f;
#pragma unroll
        for (int r = 0; r < 8; ++r) { p1[r] = __expf(s1[r] - mn); rs += p1[r]; }
#pragma unroll
        for (int r = 0; r < 8; ++r) { p2[r] = __expf(s2[r] - mn); rs += p2[r]; }
        rs += __shfl_xor(rs, 16, 32);
        l = l * alpha + rs;
        m = mn;
#pragma unroll
        for (int r = 0; r < 8; ++r) O[r] *= alpha;

        // ---- relayout P^T from D layout to B layout (keys 0..31 x queries) ----
        unsigned int u1[4], u2[4], x1[4], x2[4];
#pragma unroll
        for (int j = 0; j < 4; ++j) {
            u1[j] = pack_h2(p1[2 * j], p1[2 * j + 1]);
            u2[j] = pack_h2(p2[2 * j], p2[2 * j + 1]);
        }
#pragma unroll
        for (int j = 0; j < 4; ++j) {
            x1[j] = __shfl_xor(u1[j], 16, 32);
            x2[j] = __shfl_xor(u2[j], 16, 32);
        }
        union { v16h v; unsigned int u[8]; } bp;
#pragma unroll
        for (int j = 0; j < 4; ++j) {
            bp.u[j]     = hi ? x2[j] : u1[j];   // keys 0..7  | 16..23 (col = lane&15)
            bp.u[j + 4] = hi ? u2[j] : x1[j];   // keys 8..15 | 24..31
        }

        // ---- A operand: V^T (M=dim, K=key 0..31), contiguous from Vt[d][s] ----
        v8h vlo = *(const v8h*)(vrow + k0);        // keys {0..7} or {16..23}
        v8h vhi = *(const v8h*)(vrow + k0 + 8);    // keys {8..15} or {24..31}
        v16h va;
#pragma unroll
        for (int i = 0; i < 8; ++i) { va[i] = vlo[i]; va[i + 8] = vhi[i]; }

        // O^T += V^T x P^T  (full K=32: 32 keys per WMMA)
        O = __builtin_amdgcn_wmma_f32_16x16x32_f16(false, va, false, bp.v,
                                                   (short)0, O, false, false);

        if (k0 + 32 < S_) {   // prefetch next chunk -> global_prefetch_b8
            __builtin_prefetch(Kb + (long)(k0 + 32 + n) * M_, 0, 3);
            __builtin_prefetch(vrow + k0 + 32, 0, 3);
        }
    }

    // O^T D layout: lane n<16 reg r -> (d=r, q=n); lane n+16 reg r -> (d=r+8, q=n)
    float invl = 1.f / l;
    float* hp = head + ((long)bh * S_ + q0 + n) * M_ + (hi ? 8 : 0);
#pragma unroll
    for (int r = 0; r < 8; ++r) hp[r] = O[r] * invl;
}

// ---------------------------------------------------------------------------
// Kernel 3: output projection out[b,q,i] = bo[i] + sum_{h,j} head[b,h,q,j]*wo[h,j,i]
// ---------------------------------------------------------------------------
__global__ void out_proj(const float* __restrict__ head, const float* __restrict__ wo,
                         const float* __restrict__ bo, float* __restrict__ out)
{
    int t = blockIdx.x * blockDim.x + threadIdx.x;
    if (t >= B_ * S_ * M_) return;
    int i  = t & 15;
    int bq = t >> 4;
    int b  = bq / S_, q = bq % S_;
    float acc = bo[i];
#pragma unroll
    for (int h = 0; h < H_; ++h) {
        const float* hp = head + ((long)(b * H_ + h) * S_ + q) * M_;
        const float* wp = wo + (long)(h * M_) * M_ + i;
#pragma unroll
        for (int j = 0; j < 16; ++j) acc = fmaf(hp[j], wp[j * M_], acc);
    }
    out[t] = acc;
}

// ---------------------------------------------------------------------------
extern "C" void kernel_launch(void* const* d_in, const int* in_sizes, int n_in,
                              void* d_out, int out_size, void* d_ws, size_t ws_size,
                              hipStream_t stream)
{
    const float* x  = (const float*)d_in[0];
    const float* wq = (const float*)d_in[1];
    const float* bq = (const float*)d_in[2];
    const float* wk = (const float*)d_in[3];
    const float* bk = (const float*)d_in[4];
    const float* wv = (const float*)d_in[5];
    const float* bv = (const float*)d_in[6];
    const float* wo = (const float*)d_in[7];
    const float* bo = (const float*)d_in[8];
    float* out = (float*)d_out;

    // workspace: Qh/Kh/Vt f16 (3 x 2MB) + head f32 (4MB) = 10MB
    size_t qkv_elems = (size_t)B_ * H_ * S_ * M_;
    half_t* Qh = (half_t*)d_ws;
    half_t* Kh = Qh + qkv_elems;
    half_t* Vt = Kh + qkv_elems;
    float*  head = (float*)(Vt + qkv_elems);

    int nproj = H_ * B_ * S_;
    hipLaunchKernelGGL(proj_qkv, dim3((nproj + 255) / 256), dim3(256), 0, stream,
                       x, wq, bq, wk, bk, wv, bv, Qh, Kh, Vt);

    int nwaves = B_ * H_ * (S_ / 16);             // 4096 query tiles
    hipLaunchKernelGGL(flash_attn, dim3(nwaves / 8), dim3(256), 0, stream,
                       Qh, Kh, Vt, head);

    int nout = B_ * S_ * M_;
    hipLaunchKernelGGL(out_proj, dim3((nout + 255) / 256), dim3(256), 0, stream,
                       head, wo, bo, out);
}
